// GRU_53334903882079
// MI455X (gfx1250) — matrix operand order repaired and checked by reading, not compiled, not run on hardware
//
#include <hip/hip_runtime.h>
#include <hip/hip_bf16.h>

// ---- problem constants (match reference) ----
#define Bsz   2048
#define Tlen  512
#define Hdim  512
#define NPRED 12
#define TH3   1536   // 3*H

typedef __attribute__((ext_vector_type(16))) __bf16 v16bf;
typedef __attribute__((ext_vector_type(8)))  float  v8f;

union FragB16 { v16bf v; uint4 q[2]; };

// One-time conversion of W_hh (fp32 row-major [3H, H]) -> bf16 in scratch.
__global__ void prep_bf16_kernel(const float* __restrict__ w,
                                 __bf16* __restrict__ o, int n) {
  int i = blockIdx.x * blockDim.x + threadIdx.x;
  if (i < n) o[i] = (__bf16)w[i];
}

__device__ __forceinline__ float sigmoid_fast(float x) {
  return 1.0f / (1.0f + __expf(-x));          // v_exp_f32
}
__device__ __forceinline__ float tanh_fast(float x) {
  return 1.0f - 2.0f / (__expf(2.0f * x) + 1.0f);
}

// Persistent GRU: one workgroup owns 16 batch rows for encoder + decoder.
// 8 waves split the 96 N-tiles; each wave runs k-outer with 12 live
// accumulators so every A fragment (h) is loaded from LDS once and reused
// across 12 independent WMMA chains.
__global__ void __launch_bounds__(256, 1)
gru_persistent_kernel(const float* __restrict__ x,     // [B, T]
                      const float* __restrict__ wih,   // [3H]
                      const float* __restrict__ bih,   // [3H]
                      const float* __restrict__ bhh,   // [3H]
                      const float* __restrict__ fcw,   // [H]
                      const float* __restrict__ fcb,   // [1]
                      const __bf16* __restrict__ whh,  // [3H, H] bf16
                      float* __restrict__ out)         // [B, PRED]
{
  extern __shared__ char smem[];
  float*  ghs = (float*)smem;                                   // 16 x 1536 f32
  float*  hfs = (float*)(smem + 16*TH3*4);                      // 16 x 512  f32
  __bf16* hbs = (__bf16*)(smem + 16*TH3*4 + 16*Hdim*4);         // 16 x 512  bf16
  float*  xcs = (float*)(smem + 16*TH3*4 + 16*Hdim*4 + 16*Hdim*2); // 16 f32
  float*  wls = xcs + 16;        // wih table  [1536]
  float*  bils = wls + TH3;      // bih table  [1536]
  float*  bhls = bils + TH3;     // bhh table  [1536]
  float*  fwls = bhls + TH3;     // fc_W       [512]

  const int tid  = threadIdx.x;
  const int wave = tid >> 5;        // 8 waves (wave32)
  const int lane = tid & 31;
  const int m    = lane & 15;       // row within 16x16 tile
  const int kh   = lane >> 4;       // K-half selector per ISA 16-bit layouts
  const int row0 = blockIdx.x * 16; // batch rows owned by this WG

  // h0 = 0; preload small gate tables into LDS (global is touched afterwards
  // only for W_hh fragments and the 16 x scalars per step)
  for (int e = tid; e < 16 * Hdim; e += 256) {
    hfs[e] = 0.0f;
    hbs[e] = (__bf16)0.0f;
  }
  for (int e = tid; e < TH3; e += 256) {
    wls[e]  = wih[e];
    bils[e] = bih[e];
    bhls[e] = bhh[e];
  }
  for (int e = tid; e < Hdim; e += 256) fwls[e] = fcw[e];

  // gh[0:16, 0:1536] = hbs @ whh^T   (wave handles 12 N-tiles of 16)
  auto gemm = [&]() {
    v8f acc[12];
    #pragma unroll
    for (int ni = 0; ni < 12; ++ni)
      #pragma unroll
      for (int v = 0; v < 8; ++v) acc[ni][v] = 0.0f;

    // A fragment: lane holds h row m, two 8-elem chunks at kh*8 / 16+kh*8
    const __bf16* arow  = hbs + m * Hdim + kh * 8;
    // B fragments: lane holds W_hh row (n0+m), two 8-elem chunks at kh*8 / 16+kh*8
    const __bf16* wbase = whh + (size_t)(wave * 12 * 16 + m) * Hdim + kh * 8;

    #pragma unroll 1
    for (int k0 = 0; k0 < Hdim; k0 += 32) {
      FragB16 a;
      a.q[0] = *(const uint4*)(arow + k0);        // K = k0 +  kh*8 .. +8
      a.q[1] = *(const uint4*)(arow + k0 + 16);   // K = k0+16+kh*8 .. +8
      #pragma unroll
      for (int ni = 0; ni < 12; ++ni) {
        FragB16 w;
        const __bf16* wrow = wbase + (size_t)ni * 16 * Hdim + k0;
        w.q[0] = *(const uint4*)(wrow);           // K = k0 +  kh*8 .. +8
        w.q[1] = *(const uint4*)(wrow + 16);      // K = k0+16+kh*8 .. +8
        acc[ni] = __builtin_amdgcn_wmma_f32_16x16x32_bf16(
            /*neg_a=*/false, a.v, /*neg_b=*/false, w.v,
            /*c_mod=*/(short)0, acc[ni], /*reuse_a=*/false, /*reuse_b=*/false);
      }
    }
    // C/D layout: VGPR v -> M = v + 8*kh, N = lane&15
    #pragma unroll
    for (int ni = 0; ni < 12; ++ni) {
      const int n0 = (wave * 12 + ni) * 16;
      #pragma unroll
      for (int v = 0; v < 8; ++v)
        ghs[(v + 8 * kh) * TH3 + n0 + m] = acc[ni][v];
    }
  };

  // GRU gate combine: reads ghs/xcs/hfs + LDS tables, writes hfs + hbs
  auto combine = [&]() {
    #pragma unroll 1
    for (int e = tid; e < 16 * Hdim; e += 256) {
      const int b = e >> 9;             // H == 512
      const int j = e & (Hdim - 1);
      const float xv  = xcs[b];
      const float ghr = ghs[b * TH3 + j]          + bhls[j];
      const float ghz = ghs[b * TH3 + Hdim + j]   + bhls[Hdim + j];
      const float ghn = ghs[b * TH3 + 2*Hdim + j] + bhls[2*Hdim + j];
      const float ir  = xv * wls[j]            + bils[j];
      const float iz  = xv * wls[Hdim + j]     + bils[Hdim + j];
      const float inn = xv * wls[2*Hdim + j]   + bils[2*Hdim + j];
      const float r = sigmoid_fast(ir + ghr);
      const float z = sigmoid_fast(iz + ghz);
      const float n = tanh_fast(inn + r * ghn);
      const float hp = hfs[e];
      const float hn = (1.0f - z) * n + z * hp;
      hfs[e] = hn;
      hbs[e] = (__bf16)hn;
    }
  };

  // ---- encoder: scan over T ----
  for (int t = 0; t < Tlen; ++t) {
    __syncthreads();                      // prev combine done -> hbs valid
    if (tid < 16) xcs[tid] = x[(size_t)(row0 + tid) * Tlen + t];
    gemm();
    __syncthreads();                      // ghs + xcs ready
    combine();
  }

  // ---- decoder: out = fc(h); h = gru(h, out) ----
  const float fb = fcb[0];
  for (int p = 0; p < NPRED; ++p) {
    __syncthreads();                      // hfs/hbs valid
    if (tid < 16) {
      float s = fb;
      const float* hr = hfs + tid * Hdim;
      #pragma unroll 8
      for (int k = 0; k < Hdim; ++k) s += hr[k] * fwls[k];
      xcs[tid] = s;
      out[(size_t)(row0 + tid) * NPRED + p] = s;
    }
    gemm();                               // uses hbs (pre-update h), correct order
    __syncthreads();
    combine();
  }
}

extern "C" void kernel_launch(void* const* d_in, const int* in_sizes, int n_in,
                              void* d_out, int out_size, void* d_ws, size_t ws_size,
                              hipStream_t stream) {
  const float* x   = (const float*)d_in[0];
  const float* wih = (const float*)d_in[1];
  const float* whh = (const float*)d_in[2];
  const float* bih = (const float*)d_in[3];
  const float* bhh = (const float*)d_in[4];
  const float* fcw = (const float*)d_in[5];
  const float* fcb = (const float*)d_in[6];
  float* out = (float*)d_out;

  // W_hh fp32 -> bf16 into scratch (stays hot in the 192MB L2)
  __bf16* whh_bf = (__bf16*)d_ws;
  const int nW = TH3 * Hdim;
  prep_bf16_kernel<<<(nW + 255) / 256, 256, 0, stream>>>(whh, whh_bf, nW);

  const size_t lds = (size_t)16*TH3*4 + 16*Hdim*4 + 16*Hdim*2 + 16*4
                   + (size_t)3*TH3*4 + Hdim*4;    // ~168 KB (< 320 KB/WGP)
  gru_persistent_kernel<<<Bsz / 16, 256, lds, stream>>>(
      x, wih, bih, bhh, fcw, fcb, whh_bf, out);

  (void)in_sizes; (void)n_in; (void)out_size; (void)ws_size;
}
